// Bridge_61538291417809
// MI455X (gfx1250) — compile-verified, matrix operand order checked by reading, not run on hardware
//
#include <hip/hip_runtime.h>
#include <hip/hip_bf16.h>

// ---------------------------------------------------------------------------
// Problem constants (reference: B=2, S=2048, D_SMALL=2048, D_BIG=5120,
// I_SDM=13824, TOP_K=128). N_TOK = B*S = 4096. All dims divisible by tiles.
// ---------------------------------------------------------------------------
#define N_TOK   4096
#define D_SMALL 2048
#define D_BIG   5120
#define I_SDM   13824
#define TOPK    128
#define LN_EPS  1e-5f

typedef __attribute__((ext_vector_type(16))) __bf16 v16bf;
typedef __attribute__((ext_vector_type(2)))  __bf16 v2bf;
typedef __attribute__((ext_vector_type(8)))  float  v8f;
typedef __attribute__((ext_vector_type(4)))  int    v4i;

#define AS_GLOBAL __attribute__((address_space(1)))
#define AS_LDS    __attribute__((address_space(3)))

// CDNA5 async global->LDS copy path (ASYNCcnt), with graceful fallback.
#if defined(__gfx1250__) && __has_builtin(__builtin_amdgcn_global_load_async_to_lds_b128)
#define HAVE_ASYNC_LDS 1
#else
#define HAVE_ASYNC_LDS 0
#endif

// Native hardware converts (v_cvt_*_bf16_f32): let ISel pick the packed form.
__device__ __forceinline__ unsigned pack2bf(float lo, float hi) {
    v2bf p = { (__bf16)lo, (__bf16)hi };
    return __builtin_bit_cast(unsigned, p);
}
__device__ __forceinline__ unsigned short f2bfb(float f) {
    return __builtin_bit_cast(unsigned short, (__bf16)f);
}
__device__ __forceinline__ float bfb2f(unsigned short b) {
    return __uint_as_float(((unsigned)b) << 16);
}

// ---------------------------------------------------------------------------
// Kernel 1: LayerNorm over D_SMALL, emit bf16 activations (A-matrix of GEMM1)
// one block per token, 256 threads, 8 contiguous elements per thread.
// ---------------------------------------------------------------------------
__global__ __launch_bounds__(256) void ln_bf16_kernel(
    const float* __restrict__ h, const float* __restrict__ sc,
    const float* __restrict__ bi, unsigned short* __restrict__ xbf)
{
    const int n   = blockIdx.x;
    const int tid = threadIdx.x;
    const float* row = h + (size_t)n * D_SMALL;

    __shared__ float red[256];

    float v[8];
    const float4* rp = reinterpret_cast<const float4*>(row + tid * 8);
    float4 a = rp[0], b = rp[1];
    v[0]=a.x; v[1]=a.y; v[2]=a.z; v[3]=a.w; v[4]=b.x; v[5]=b.y; v[6]=b.z; v[7]=b.w;

    float s = 0.f;
#pragma unroll
    for (int i = 0; i < 8; ++i) s += v[i];
    red[tid] = s; __syncthreads();
    for (int o = 128; o > 0; o >>= 1) { if (tid < o) red[tid] += red[tid + o]; __syncthreads(); }
    const float mu = red[0] * (1.f / D_SMALL);
    __syncthreads();

    float ss = 0.f;
#pragma unroll
    for (int i = 0; i < 8; ++i) { float d = v[i] - mu; ss += d * d; }
    red[tid] = ss; __syncthreads();
    for (int o = 128; o > 0; o >>= 1) { if (tid < o) red[tid] += red[tid + o]; __syncthreads(); }
    const float rstd = rsqrtf(red[0] * (1.f / D_SMALL) + LN_EPS);

    unsigned pk[4];
#pragma unroll
    for (int i = 0; i < 4; ++i) {
        const int c = tid * 8 + 2 * i;
        float y0 = (v[2*i]   - mu) * rstd * sc[c]   + bi[c];
        float y1 = (v[2*i+1] - mu) * rstd * sc[c+1] + bi[c+1];
        pk[i] = pack2bf(y0, y1);
    }
    uint4* dst = reinterpret_cast<uint4*>(xbf + (size_t)n * D_SMALL + tid * 8);
    *dst = make_uint4(pk[0], pk[1], pk[2], pk[3]);
}

// ---------------------------------------------------------------------------
// Kernel 2: templated WMMA-bf16 GEMM:  C[M,N] = A(bf16,[M,K]) @ W(f32,[N,K])^T
// CTA tile 128x128x64, 8 waves (wave32), each wave = 32(M) x 64(N) = 2x4 frags
// of v_wmma_f32_16x16x32_bf16.
//
// Software pipeline with double-buffered LDS (64 KB total):
//   - A tile (already bf16) is copied with GLOBAL_LOAD_ASYNC_TO_LDS_B128
//     (ASYNCcnt path), issued for stage k+1 before computing stage k.
//   - W tile (f32 in HBM, the unavoidable one-pass read) is loaded into
//     registers pre-compute and converted f32->bf16 + ds_store post-compute,
//     so s_wait_loadcnt lands behind the 16 WMMAs of the current stage.
// Epilogues: 0 = store bf16, 1 = silu->bf16, 2 = residual: out=h+c*tanh(gs[n])
// ---------------------------------------------------------------------------
#define BM 128
#define BN 128
#define BK 64

template <int EPI>
__global__ __launch_bounds__(256) void gemm_wmma_bf16(
    const unsigned short* __restrict__ A,   // [M,K] bf16
    const float*          __restrict__ W,   // [N,K] f32
    unsigned short*       __restrict__ ObF, // [M,N] bf16 (EPI 0/1)
    float*                __restrict__ OF,  // [M,N] f32  (EPI 2)
    const float*          __restrict__ Hres,// [M,N] f32  (EPI 2)
    const float*          __restrict__ gsc, // [N]        (EPI 2)
    int M, int N, int K)
{
    __shared__ __align__(32) unsigned short sA[2][BM * BK]; // 2 x 16 KB
    __shared__ __align__(32) unsigned short sB[2][BN * BK]; // 2 x 16 KB

    const int tid  = threadIdx.x;
    const int lane = tid & 31;
    const int wave = tid >> 5;
    const int wm   = wave & 3;   // M offset = 32*wm
    const int wn   = wave >> 2;  // N offset = 64*wn

    const int m0 = blockIdx.y * BM;
    const int n0 = blockIdx.x * BN;

    const int lrow = tid >> 1;   // 0..127 : tile row loaded by this thread
    const int lseg = tid & 1;    // 0..1   : 32-element K segment

    const size_t aRowBase = (size_t)(m0 + lrow) * K + lseg * 32;
    const size_t wRowBase = (size_t)(n0 + lrow) * K + lseg * 32;
    const int    ldsOff   = lrow * BK + lseg * 32;

    v8f acc[2][4];
    const v8f vzero = {};
#pragma unroll
    for (int mf = 0; mf < 2; ++mf)
#pragma unroll
        for (int nf = 0; nf < 4; ++nf) acc[mf][nf] = vzero;

    float4 breg[8]; // staged W values for the next K-slab

    // ---- stage-A issue: async copy global(bf16) -> LDS, 32B per thread ----
    auto issueA = [&](int buf, int k0) {
#if HAVE_ASYNC_LDS
        const unsigned short* gp = A + aRowBase + k0;
        unsigned short*       lp = &sA[buf][ldsOff];
        __builtin_amdgcn_global_load_async_to_lds_b128(
            (AS_GLOBAL v4i*)gp, (AS_LDS v4i*)lp, 0, 0);
        __builtin_amdgcn_global_load_async_to_lds_b128(
            (AS_GLOBAL v4i*)(gp + 8), (AS_LDS v4i*)(lp + 8), 0, 0);
#else
        const uint4* ap = reinterpret_cast<const uint4*>(A + aRowBase + k0);
        uint4* da = reinterpret_cast<uint4*>(&sA[buf][ldsOff]);
        da[0] = ap[0]; da[1] = ap[1];
#endif
    };
    auto waitAsync = [&]() {
#if HAVE_ASYNC_LDS
#if __has_builtin(__builtin_amdgcn_s_wait_asynccnt)
        __builtin_amdgcn_s_wait_asynccnt(0);
#else
        asm volatile("s_wait_asynccnt 0x0" ::: "memory");
#endif
#endif
    };
    // ---- stage-B issue: f32 loads into registers (wait deferred to use) ----
    auto issueB = [&](int k0) {
        const float4* wp = reinterpret_cast<const float4*>(W + wRowBase + k0);
#pragma unroll
        for (int i = 0; i < 8; ++i) breg[i] = wp[i];
    };
    // ---- stage-B commit: convert f32->bf16 (packed cvt), store 64B to LDS ----
    auto storeB = [&](int buf) {
        uint4 pk[4];
        unsigned* pw = reinterpret_cast<unsigned*>(pk);
#pragma unroll
        for (int i = 0; i < 8; ++i) {
            pw[2*i]   = pack2bf(breg[i].x, breg[i].y);
            pw[2*i+1] = pack2bf(breg[i].z, breg[i].w);
        }
        uint4* db = reinterpret_cast<uint4*>(&sB[buf][ldsOff]);
        db[0] = pk[0]; db[1] = pk[1]; db[2] = pk[2]; db[3] = pk[3];
    };
    // ---- compute one K-slab from LDS buffer `buf`: 2 k-steps x 8 WMMA ----
    auto compute = [&](int buf) {
#pragma unroll
        for (int ks = 0; ks < 2; ++ks) {
            const int kloc = ks * 32 + ((lane >> 4) << 4); // lanes 16..31: K+16..31
            v16bf afr[2], bfr[4];
#pragma unroll
            for (int mf = 0; mf < 2; ++mf)
                afr[mf] = *reinterpret_cast<const v16bf*>(
                    &sA[buf][(wm * 32 + mf * 16 + (lane & 15)) * BK + kloc]);
#pragma unroll
            for (int nf = 0; nf < 4; ++nf)
                bfr[nf] = *reinterpret_cast<const v16bf*>(
                    &sB[buf][(wn * 64 + nf * 16 + (lane & 15)) * BK + kloc]);
#pragma unroll
            for (int mf = 0; mf < 2; ++mf)
#pragma unroll
                for (int nf = 0; nf < 4; ++nf)
                    acc[mf][nf] = __builtin_amdgcn_wmma_f32_16x16x32_bf16(
                        false, afr[mf], false, bfr[nf],
                        (short)0, acc[mf][nf], false, false);
        }
    };

    // ---- pipeline prologue: fill buffer 0 ----
    issueA(0, 0);
    issueB(0);
    storeB(0);
    waitAsync();
    __syncthreads();

    // ---- steady state ----
    int cur = 0;
    for (int k0 = 0; k0 < K; k0 += BK) {
        const bool hasNext = (k0 + BK) < K;
        if (hasNext) {
            issueA(cur ^ 1, k0 + BK);   // async: lands in LDS during compute
            issueB(k0 + BK);            // in flight: wait deferred to storeB
        }
        compute(cur);                   // 16x v_wmma hide the load latency
        if (hasNext) {
            storeB(cur ^ 1);            // s_wait_loadcnt + cvt + ds_store here
            waitAsync();                // A tile complete before publishing
        }
        __syncthreads();
        cur ^= 1;
    }

    // ---- epilogue: C frag layout: VGPR r -> M = r + 8*(lane>=16), N = lane&15
#pragma unroll
    for (int mf = 0; mf < 2; ++mf) {
#pragma unroll
        for (int nf = 0; nf < 4; ++nf) {
#pragma unroll
            for (int r = 0; r < 8; ++r) {
                const int m = m0 + wm * 32 + mf * 16 + r + ((lane >> 4) << 3);
                const int n = n0 + wn * 64 + nf * 16 + (lane & 15);
                const float v = acc[mf][nf][r];
                if (EPI == 0) {
                    ObF[(size_t)m * N + n] = f2bfb(v);
                } else if (EPI == 1) {
                    const float s = v / (1.f + __expf(-v));   // silu
                    ObF[(size_t)m * N + n] = f2bfb(s);
                } else {
                    OF[(size_t)m * N + n] =
                        Hres[(size_t)m * N + n] + v * tanhf(gsc[n]);
                }
            }
        }
    }
}

// ---------------------------------------------------------------------------
// Kernel 3: per-token top-k(128) by |silu(gate)| via 4096-bin radix select on
// bf16 magnitude bits, fused with the scatter that materializes the gu_dense
// row (zeros elsewhere). bf16 bits (sign cleared) are monotone in |value|, so
// bin = (bits & 0x7FFF) >> 3 ranks directly. One block per token.
// ---------------------------------------------------------------------------
__global__ __launch_bounds__(256) void topk_scatter_kernel(
    const unsigned short* __restrict__ g,       // [N_TOK, I_SDM] silu(gate), bf16
    const unsigned short* __restrict__ u,       // [N_TOK, I_SDM] up,         bf16
    unsigned short*       __restrict__ gu_dense)// [N_TOK, I_SDM] out,        bf16
{
    const int n   = blockIdx.x;
    const int tid = threadIdx.x;
    const unsigned short* grow = g + (size_t)n * I_SDM;
    const unsigned short* urow = u + (size_t)n * I_SDM;
    unsigned short*       orow = gu_dense + (size_t)n * I_SDM;

    __shared__ int hist[4096];
    __shared__ int s_bstar, s_rem, s_eqcnt;

    for (int i = tid; i < 4096; i += 256) hist[i] = 0;
    if (tid == 0) s_eqcnt = 0;
    __syncthreads();

    for (int j = tid; j < I_SDM; j += 256) {
        const int bin = (grow[j] & 0x7FFF) >> 3;
        atomicAdd(&hist[bin], 1);
    }
    __syncthreads();

    if (tid == 0) {
        int cum = 0, b = 4095;
        for (; b > 0; --b) { cum += hist[b]; if (cum >= TOPK) break; }
        if (cum < TOPK) cum += hist[0];            // b == 0 fallthrough
        s_bstar = b;
        s_rem   = TOPK - (cum - hist[b]);          // how many to take from bin b
    }
    __syncthreads();
    const int bstar = s_bstar, rem = s_rem;

    for (int j = tid; j < I_SDM; j += 256) {
        const unsigned short gb = grow[j];
        const int bin = (gb & 0x7FFF) >> 3;
        bool take = (bin > bstar);
        if (!take && bin == bstar) {
            const int r = atomicAdd(&s_eqcnt, 1);
            take = (r < rem);
        }
        unsigned short out = 0;
        if (take) out = f2bfb(bfb2f(gb) * bfb2f(urow[j]));
        orow[j] = out;                              // zero or g*u -> full row written
    }
}

// ---------------------------------------------------------------------------
// Host orchestration.
// Workspace layout (peak ~399 MB):
//   x_bf   [4096,2048]  bf16   16 MB
//   xb_bf  [4096,5120]  bf16   40 MB
//   g_bf   [4096,13824] bf16  108 MB   (reused for rb_bf after top-k)
//   u_bf   [4096,13824] bf16  108 MB
//   gu_bf  [4096,13824] bf16  108 MB
// ---------------------------------------------------------------------------
extern "C" void kernel_launch(void* const* d_in, const int* in_sizes, int n_in,
                              void* d_out, int out_size, void* d_ws, size_t ws_size,
                              hipStream_t stream) {
    (void)in_sizes; (void)n_in; (void)out_size; (void)ws_size;

    const float* h          = (const float*)d_in[0];
    const float* ln_scale   = (const float*)d_in[1];
    const float* ln_bias    = (const float*)d_in[2];
    const float* w_in       = (const float*)d_in[3];
    const float* w_out      = (const float*)d_in[4];
    const float* gate_small = (const float*)d_in[5];
    const float* sdm_gate   = (const float*)d_in[6];
    const float* sdm_up     = (const float*)d_in[7];
    const float* sdm_down   = (const float*)d_in[8];
    float* out = (float*)d_out;

    char* ws = (char*)d_ws;
    size_t off = 0;
    auto carve = [&](size_t bytes) -> char* {
        char* p = ws + off;
        off += (bytes + 255) & ~(size_t)255;
        return p;
    };
    unsigned short* x_bf  = (unsigned short*)carve((size_t)N_TOK * D_SMALL * 2);
    unsigned short* xb_bf = (unsigned short*)carve((size_t)N_TOK * D_BIG   * 2);
    unsigned short* g_bf  = (unsigned short*)carve((size_t)N_TOK * I_SDM   * 2);
    unsigned short* u_bf  = (unsigned short*)carve((size_t)N_TOK * I_SDM   * 2);
    unsigned short* gu_bf = (unsigned short*)carve((size_t)N_TOK * I_SDM   * 2);
    unsigned short* rb_bf = g_bf;   // g dead after top-k; reuse for retrieved_big

    // 1) LayerNorm -> x (bf16)
    ln_bf16_kernel<<<N_TOK, 256, 0, stream>>>(h, ln_scale, ln_bias, x_bf);

    // 2) proj_in: xb = x @ w_in^T          [4096,5120]
    gemm_wmma_bf16<0><<<dim3(D_BIG / BN, N_TOK / BM), 256, 0, stream>>>(
        x_bf, w_in, xb_bf, nullptr, nullptr, nullptr, N_TOK, D_BIG, D_SMALL);

    // 3) gate: g = silu(xb @ sdm_gate^T)   [4096,13824]
    gemm_wmma_bf16<1><<<dim3(I_SDM / BN, N_TOK / BM), 256, 0, stream>>>(
        xb_bf, sdm_gate, g_bf, nullptr, nullptr, nullptr, N_TOK, I_SDM, D_BIG);

    // 4) up: u = xb @ sdm_up^T             [4096,13824]
    gemm_wmma_bf16<0><<<dim3(I_SDM / BN, N_TOK / BM), 256, 0, stream>>>(
        xb_bf, sdm_up, u_bf, nullptr, nullptr, nullptr, N_TOK, I_SDM, D_BIG);

    // 5) top-k select + scatter -> gu_dense (row fully written: zeros + g*u)
    topk_scatter_kernel<<<N_TOK, 256, 0, stream>>>(g_bf, u_bf, gu_bf);

    // 6) down: rb = gu_dense @ sdm_down^T  [4096,5120]  (dense: one-pass weight read)
    gemm_wmma_bf16<0><<<dim3(D_BIG / BN, N_TOK / BM), 256, 0, stream>>>(
        gu_bf, sdm_down, rb_bf, nullptr, nullptr, nullptr, N_TOK, D_BIG, I_SDM);

    // 7) proj_out + residual gate: out = h + (rb @ w_out^T) * tanh(gate_small)
    gemm_wmma_bf16<2><<<dim3(D_SMALL / BN, N_TOK / BM), 256, 0, stream>>>(
        rb_bf, w_out, nullptr, out, h, gate_small, N_TOK, D_SMALL, D_BIG);
}
